// TestModel_27556510171510
// MI455X (gfx1250) — matrix operand order repaired and checked by reading, not compiled, not run on hardware
//
#include <hip/hip_runtime.h>

#define T_LEN 32000
#define B_SZ 8
#define TB 256000            // T*B
#define LOG2E 1.44269504088896340736f
#define SCALE_X 1.41421356237309504880f

typedef unsigned short u16_t;
typedef __attribute__((ext_vector_type(16))) __bf16 v16bf;
typedef __attribute__((ext_vector_type(8)))  float  v8f;

union FragBF { unsigned int u[8]; v16bf v; };

__device__ __forceinline__ u16_t f2bf(float x) {
    unsigned int u = __builtin_bit_cast(unsigned int, x);
    unsigned int r = u + 0x7FFFu + ((u >> 16) & 1u);   // round-to-nearest-even
    return (u16_t)(r >> 16);
}

__device__ __forceinline__ float sigf(float x) {
    // sigmoid via v_exp_f32 + v_rcp_f32 (short dependent chain)
    float e = __builtin_amdgcn_exp2f(x * -LOG2E);
    return __builtin_amdgcn_rcpf(1.0f + e);
}

// ---------------------------------------------------------------------------
// prep: Mc[s][9][512] = W_first_block^T @ W0_s ; Ub[s][512] = b_first_block @ W0_s
//       W1t[s][384][128] = bf16(W1_s^T)   (column-major B operand for WMMA)
// ---------------------------------------------------------------------------
__global__ void prep_kernel(const float* __restrict__ Wf, const float* __restrict__ bf,
                            const float* __restrict__ aW0, const float* __restrict__ bW0,
                            const float* __restrict__ aW1, const float* __restrict__ bW1,
                            float* __restrict__ Mc, float* __restrict__ Ub,
                            u16_t* __restrict__ W1t) {
    int tid = threadIdx.x;
    for (int n = tid; n < 2 * 9 * 512; n += 256) {
        int j = n % 512, i = (n / 512) % 9, s = n / (9 * 512);
        const float* W0 = s ? bW0 : aW0;
        float acc = 0.f;
        for (int c = 0; c < 64; ++c) acc += Wf[(s * 64 + c) * 9 + i] * W0[c * 512 + j];
        Mc[n] = acc;
    }
    for (int n = tid; n < 1024; n += 256) {
        int j = n & 511, s = n >> 9;
        const float* W0 = s ? bW0 : aW0;
        float acc = 0.f;
        for (int c = 0; c < 64; ++c) acc += bf[s * 64 + c] * W0[c * 512 + j];
        Ub[n] = acc;
    }
    for (int n = tid; n < 2 * 384 * 128; n += 256) {
        int k = n & 127, nn = (n >> 7) % 384, s = n / (384 * 128);
        const float* W1 = s ? bW1 : aW1;
        W1t[n] = f2bf(W1[(size_t)k * 384 + nn]);
    }
}

// ---------------------------------------------------------------------------
// u0: U0[s][t][b][g*128+d] = Ub[s][j] + sum_i tensor[b][i][t] * Mc[s][i][j]
// K=9 -> VALU mini-GEMM; purely store-bandwidth bound. j coalesced.
// ---------------------------------------------------------------------------
__global__ void u0_kernel(const float* __restrict__ x, const float* __restrict__ Mc,
                          const float* __restrict__ Ub, float* __restrict__ U0) {
    long long n = (long long)blockIdx.x * blockDim.x + threadIdx.x;  // < 2*T*B*512
    int j = (int)(n & 511);
    int b = (int)((n >> 9) & 7);
    int s = (int)((n >> 12) & 1);
    int t = (int)(n >> 13);
    const float* M = Mc + s * 9 * 512;
    float acc = Ub[s * 512 + j];
    const float* xp = x + (size_t)b * 9 * T_LEN + t;
#pragma unroll
    for (int i = 0; i < 9; ++i) acc += xp[(size_t)i * T_LEN] * M[i * 512 + j];
    U0[(size_t)s * ((size_t)TB * 512) + ((size_t)t * 8 + b) * 512 + j] = acc;
}

// ---------------------------------------------------------------------------
// scan0: SRU recurrence layer 0 (k=4, res = u3). 2048 chains, 1 chain/lane.
// Writes h in fp32 (residual for layer1) and bf16 (WMMA A operand).
// ---------------------------------------------------------------------------
__global__ void scan0_kernel(const float* __restrict__ U0,
                             const float* __restrict__ av, const float* __restrict__ ab,
                             const float* __restrict__ bv, const float* __restrict__ bb,
                             u16_t* __restrict__ H0bf, float* __restrict__ H0f) {
    int s = blockIdx.x >> 3, b = blockIdx.x & 7, d = threadIdx.x;
    const float* v = s ? bv : av;
    const float* g = s ? bb : ab;
    float vf = v[d], vr = v[128 + d], bfc = g[d], brc = g[128 + d];
    const float* U = U0 + (size_t)s * ((size_t)TB * 512);
    u16_t* Hb = H0bf + (size_t)s * ((size_t)TB * 128);
    float* Hf = H0f + (size_t)s * ((size_t)TB * 128);

    size_t idx = (size_t)b * 512 + d;
    float u0 = U[idx], u1 = U[idx + 128], u2 = U[idx + 256], u3 = U[idx + 384];
    float c = 0.f;
    for (int t = 0; t < T_LEN; ++t) {
        size_t nidx = idx + 4096;
        float n0 = 0.f, n1 = 0.f, n2 = 0.f, n3 = 0.f;
        if (t + 1 < T_LEN) {                         // prefetch next step off the chain
            n0 = U[nidx]; n1 = U[nidx + 128]; n2 = U[nidx + 256]; n3 = U[nidx + 384];
            __builtin_prefetch(&U[nidx + 8 * 4096], 0, 0);   // global_prefetch_b8
        }
        float f = sigf(u1 + vf * c + bfc);
        c = u0 + f * (c - u0);
        float r = sigf(u2 + vr * c + brc);
        float rsx = u3 * SCALE_X;
        float h = rsx + r * (c - rsx);
        size_t ho = (size_t)t * 1024 + (size_t)b * 128 + d;
        Hf[ho] = h;
        Hb[ho] = f2bf(h);
        u0 = n0; u1 = n1; u2 = n2; u3 = n3;
        idx = nidx;
    }
}

// ---------------------------------------------------------------------------
// u1: U1 = H0(bf16) @ W1(bf16), M=TB, K=128, N=384, f32 accumulate.
// One 16x16 tile per wave, 4x v_wmma_f32_16x16x32_bf16. A staged via LDS.
// ---------------------------------------------------------------------------
__global__ void u1_wmma_kernel(const u16_t* __restrict__ H0bf,
                               const u16_t* __restrict__ W1t,
                               float* __restrict__ U1) {
    __shared__ unsigned int ldsA[1024];            // 16 rows x 128 bf16 = 4KB
    int s = blockIdx.z;
    int m0 = blockIdx.x * 16;
    int wave = threadIdx.x >> 5, lane = threadIdx.x & 31;
    int col = lane & 15, hi = lane >> 4;

    const u16_t* A = H0bf + (size_t)s * ((size_t)TB * 128);
    ((uint4*)ldsA)[threadIdx.x] = ((const uint4*)(A + (size_t)m0 * 128))[threadIdx.x];
    __syncthreads();

    int n0 = (blockIdx.y * 8 + wave) * 16;
    const unsigned int* Bw = (const unsigned int*)(W1t + (size_t)s * 384 * 128);

    v8f acc;
#pragma unroll
    for (int q = 0; q < 8; ++q) acc[q] = 0.f;

#pragma unroll
    for (int kc = 0; kc < 4; ++kc) {
        FragBF a, bm;
#pragma unroll
        for (int g = 0; g < 8; ++g) {
            // A (16x32 bf16): VGPR g holds K pair {0,2,4,6 / 16,18,20,22} + hi*8
            int kkA = kc * 32 + hi * 8 + ((g < 4) ? 2 * g : 16 + 2 * (g - 4));
            a.u[g] = ldsA[col * 64 + (kkA >> 1)];
            // B (32x16 bf16): same K order as A ({0,2,4,6 / 16,18,20,22} + hi*8)
            int kkB = kc * 32 + hi * 8 + ((g < 4) ? 2 * g : 16 + 2 * (g - 4));
            bm.u[g] = Bw[(size_t)(n0 + col) * 64 + (kkB >> 1)];
        }
        acc = __builtin_amdgcn_wmma_f32_16x16x32_bf16(
            false, a.v, false, bm.v, (short)0, acc, false, false);
    }

    float* Up = U1 + (size_t)s * ((size_t)TB * 384);
#pragma unroll
    for (int r = 0; r < 8; ++r) {                  // C/D: VGPR r -> M = r + 8*hi
        int M = m0 + r + hi * 8;
        Up[(size_t)M * 384 + n0 + col] = acc[r];
    }
}

// ---------------------------------------------------------------------------
// scan1: SRU recurrence layer 1 (k=3, res = h0). Writes h1 fp32.
// ---------------------------------------------------------------------------
__global__ void scan1_kernel(const float* __restrict__ U1, const float* __restrict__ H0f,
                             const float* __restrict__ av, const float* __restrict__ ab,
                             const float* __restrict__ bv, const float* __restrict__ bb,
                             float* __restrict__ H1f) {
    int s = blockIdx.x >> 3, b = blockIdx.x & 7, d = threadIdx.x;
    const float* v = s ? bv : av;
    const float* g = s ? bb : ab;
    float vf = v[d], vr = v[128 + d], bfc = g[d], brc = g[128 + d];
    const float* U = U1 + (size_t)s * ((size_t)TB * 384);
    const float* R = H0f + (size_t)s * ((size_t)TB * 128);
    float* H = H1f + (size_t)s * ((size_t)TB * 128);

    size_t idx = (size_t)b * 384 + d;
    size_t ridx = (size_t)b * 128 + d;
    float u0 = U[idx], u1 = U[idx + 128], u2 = U[idx + 256], u3 = R[ridx];
    float c = 0.f;
    for (int t = 0; t < T_LEN; ++t) {
        size_t nidx = idx + 3072;
        size_t nr = ridx + 1024;
        float n0 = 0.f, n1 = 0.f, n2 = 0.f, n3 = 0.f;
        if (t + 1 < T_LEN) {
            n0 = U[nidx]; n1 = U[nidx + 128]; n2 = U[nidx + 256]; n3 = R[nr];
            __builtin_prefetch(&U[nidx + 8 * 3072], 0, 0);
        }
        float f = sigf(u1 + vf * c + bfc);
        c = u0 + f * (c - u0);
        float r = sigf(u2 + vr * c + brc);
        float rsx = u3 * SCALE_X;
        H[ridx] = rsx + r * (c - rsx);
        u0 = n0; u1 = n1; u2 = n2; u3 = n3;
        idx = nidx; ridx = nr;
    }
}

// ---------------------------------------------------------------------------
// final: out[b][0][t] = b_last + sum_d Wl[d] * (h1a + h1b)[t][b][d]
// one wave per 8 rows, float4 lane loads + shuffle reduction
// ---------------------------------------------------------------------------
__global__ void final_kernel(const float* __restrict__ H1, const float* __restrict__ Wl,
                             const float* __restrict__ bl, float* __restrict__ out) {
    int lane = threadIdx.x & 31, wave = threadIdx.x >> 5;
    int baseRow = (blockIdx.x * (blockDim.x >> 5) + wave) * 8;
    float4 w4 = ((const float4*)Wl)[lane];
    const float* Ha = H1;
    const float* Hb = H1 + (size_t)TB * 128;
    float blv = bl[0];
#pragma unroll
    for (int o = 0; o < 8; ++o) {
        int row = baseRow + o;
        if (row >= TB) break;
        float4 a = ((const float4*)(Ha + (size_t)row * 128))[lane];
        float4 b = ((const float4*)(Hb + (size_t)row * 128))[lane];
        float p = w4.x * (a.x + b.x) + w4.y * (a.y + b.y) +
                  w4.z * (a.z + b.z) + w4.w * (a.w + b.w);
        for (int m = 16; m >= 1; m >>= 1) p += __shfl_xor(p, m, 32);
        if (lane == 0) {
            int bb = row & 7, t = row >> 3;
            out[(size_t)bb * T_LEN + t] = p + blv;
        }
    }
}

extern "C" void kernel_launch(void* const* d_in, const int* in_sizes, int n_in,
                              void* d_out, int out_size, void* d_ws, size_t ws_size,
                              hipStream_t stream) {
    const float* tensor  = (const float*)d_in[0];
    const float* W_first = (const float*)d_in[1];
    const float* b_first = (const float*)d_in[2];
    const float* W_last  = (const float*)d_in[3];
    const float* b_last  = (const float*)d_in[4];
    const float* a_W0 = (const float*)d_in[5];
    const float* a_v0 = (const float*)d_in[6];
    const float* a_b0 = (const float*)d_in[7];
    const float* a_W1 = (const float*)d_in[8];
    const float* a_v1 = (const float*)d_in[9];
    const float* a_b1 = (const float*)d_in[10];
    const float* b_W0 = (const float*)d_in[11];
    const float* b_v0 = (const float*)d_in[12];
    const float* b_b0 = (const float*)d_in[13];
    const float* b_W1 = (const float*)d_in[14];
    const float* b_v1 = (const float*)d_in[15];
    const float* b_b1 = (const float*)d_in[16];
    float* out = (float*)d_out;

    char* ws = (char*)d_ws;
    float* Mc   = (float*)(ws + 0);                         // 2*9*512 f32
    float* Ub   = (float*)(ws + 36864);                     // 2*512 f32
    u16_t* W1t  = (u16_t*)(ws + 40960);                     // 2*384*128 bf16
    size_t off_U   = 262144;
    size_t szU0   = (size_t)2 * TB * 512 * 4;               // 1,048,576,000 (U1 reuses)
    size_t off_H0b = off_U + szU0;
    size_t szH0b  = (size_t)2 * TB * 128 * 2;               // 131,072,000
    size_t off_H0f = off_H0b + szH0b;
    size_t szH0f  = (size_t)2 * TB * 128 * 4;               // 262,144,000
    size_t off_H1  = off_H0f + szH0f;
    float* U0  = (float*)(ws + off_U);
    float* U1  = (float*)(ws + off_U);                      // overwrite after scan0
    u16_t* H0b = (u16_t*)(ws + off_H0b);
    float* H0f = (float*)(ws + off_H0f);
    float* H1f = (float*)(ws + off_H1);

    prep_kernel<<<1, 256, 0, stream>>>(W_first, b_first, a_W0, b_W0, a_W1, b_W1,
                                       Mc, Ub, W1t);

    long long totU0 = (long long)2 * TB * 512;              // 262,144,000
    u0_kernel<<<(unsigned)(totU0 / 256), 256, 0, stream>>>(tensor, Mc, Ub, U0);

    scan0_kernel<<<16, 128, 0, stream>>>(U0, a_v0, a_b0, b_v0, b_b0, H0b, H0f);

    dim3 g1(TB / 16, 3, 2);                                 // 16000 x 3 x 2
    u1_wmma_kernel<<<g1, 256, 0, stream>>>(H0b, W1t, U1);

    scan1_kernel<<<16, 128, 0, stream>>>(U1, H0f, a_v1, a_b1, b_v1, b_b1, H1f);

    final_kernel<<<TB / 64, 256, 0, stream>>>(H1f, W_last, b_last, out);
}
